// LookupKAN2D_Residual_Efficient_2293512536825
// MI455X (gfx1250) — compile-verified, matrix operand order checked by reading, not run on hardware
//
#include <hip/hip_runtime.h>

// Problem constants (from reference)
#define BATCH   4096
#define PI_     16      // input pairs
#define HID     64
#define PH_     32      // hidden pairs
#define OUT     32
#define NG      32      // num grids
// Table strides (floats)
// L0[gi,gj,h,p]: gi:33792  gj:1024  h:16  p:1
// L1[gi,gj,o,p,z]: gi:540672  gj:16384  o:512  p:32  z:1
#define L0_GI 33792
#define L0_GJ 1024
#define L1_GI 540672
#define L1_GJ 16384
#define L1_O  512
#define L1_P  32

typedef __attribute__((ext_vector_type(16))) _Float16 v16h;
typedef __attribute__((ext_vector_type(8)))  float    v8f;

__device__ __forceinline__ void laplace_interp(float x, const float* sb, const float* sinv,
                                               int& i, float& d) {
  float e = expf(-fabsf(x));
  float c = (x > 0.f) ? (1.f - 0.5f * e) : (0.5f * e);
  float s = c * 32.f;
  int ii = (int)s;                       // s >= 0 -> trunc == floor
  ii = ii < 0 ? 0 : (ii > NG - 1 ? NG - 1 : ii);
  d = (x - sb[ii]) * sinv[ii];
  i = ii;
}

// ---------------- Kernel 1: per-(p,b) level-0 interp indices/weights ----------------
__global__ void k_interp_x(const float* __restrict__ x,
                           const float* __restrict__ borders,
                           const float* __restrict__ invlen,
                           int* __restrict__ I1, int* __restrict__ I2,
                           float* __restrict__ W00, float* __restrict__ W01,
                           float* __restrict__ W10, float* __restrict__ W11) {
  __shared__ float sb[33];
  __shared__ float sinv[32];
  int t = threadIdx.x;
  if (t < 33) sb[t] = borders[t];
  if (t < 32) sinv[t] = invlen[t];
  __syncthreads();
  int tid = blockIdx.x * blockDim.x + t;          // tid = p*BATCH + b
  int p = tid >> 12;
  int b = tid & (BATCH - 1);
  float x1 = x[(2 * p) * BATCH + b];
  float x2 = x[(2 * p + 1) * BATCH + b];
  int i1, i2; float d1, d2;
  laplace_interp(x1, sb, sinv, i1, d1);
  laplace_interp(x2, sb, sinv, i2, d2);
  I1[tid] = i1;
  I2[tid] = i2;
  W00[tid] = (1.f - d1) * (1.f - d2);
  W01[tid] = (1.f - d1) * d2;
  W10[tid] = d1 * (1.f - d2);
  W11[tid] = d1 * d2;
}

// ---------------- Kernel 2: z0[h,b] = 4-corner gather-reduce over L0 ----------------
__global__ void k_level0(const float* __restrict__ L0,
                         const int* __restrict__ I1, const int* __restrict__ I2,
                         const float* __restrict__ W00, const float* __restrict__ W01,
                         const float* __restrict__ W10, const float* __restrict__ W11,
                         float* __restrict__ Z0) {
  int tid = blockIdx.x * blockDim.x + threadIdx.x; // tid = h*BATCH + b
  int h = tid >> 12;
  int b = tid & (BATCH - 1);
  float acc = 0.f;
#pragma unroll
  for (int p = 0; p < PI_; ++p) {
    int q = p * BATCH + b;
    int i1 = I1[q];
    int i2 = I2[q];
    float w00 = W00[q], w01 = W01[q], w10 = W10[q], w11 = W11[q];
    int base = i1 * L0_GI + i2 * L0_GJ + h * 16 + p;
    float a00 = L0[base];
    float a01 = L0[base + L0_GJ];
    float a10 = L0[base + L0_GI];
    float a11 = L0[base + L0_GI + L0_GJ];
    acc = fmaf(w00, a00, acc);
    acc = fmaf(w01, a01, acc);
    acc = fmaf(w10, a10, acc);
    acc = fmaf(w11, a11, acc);
  }
  Z0[tid] = acc;
}

// ---------------- Kernel 3: per-(z,b) level-1 interp (only J1, wz00, wz10 needed) ----
__global__ void k_interp_z(const float* __restrict__ Z0,
                           const float* __restrict__ borders,
                           const float* __restrict__ invlen,
                           int* __restrict__ J1,
                           float* __restrict__ WZ00, float* __restrict__ WZ10) {
  __shared__ float sb[33];
  __shared__ float sinv[32];
  int t = threadIdx.x;
  if (t < 33) sb[t] = borders[t];
  if (t < 32) sinv[t] = invlen[t];
  __syncthreads();
  int tid = blockIdx.x * blockDim.x + t;          // tid = z*BATCH + b
  int z = tid >> 12;
  int b = tid & (BATCH - 1);
  float z1 = Z0[(2 * z) * BATCH + b];
  float z2 = Z0[(2 * z + 1) * BATCH + b];
  int j1, j2; float e1, e2;
  laplace_interp(z1, sb, sinv, j1, e1);
  laplace_interp(z2, sb, sinv, j2, e2);
  J1[tid] = j1;
  WZ00[tid] = (1.f - e1) * (1.f - e2);
  WZ10[tid] = e1 * (1.f - e2);
}

// ---------------- Kernel 4: out[o,b] gather-reduce over 71MB L1 (L2-resident) -------
__global__ void k_level1(const float* __restrict__ L1,
                         const int* __restrict__ I1,
                         const float* __restrict__ W00, const float* __restrict__ W10,
                         const int* __restrict__ J1,
                         const float* __restrict__ WZ00, const float* __restrict__ WZ10,
                         float* __restrict__ out) {
  int tid = blockIdx.x * blockDim.x + threadIdx.x; // tid = o*BATCH + b
  int o = tid >> 12;
  int b = tid & (BATCH - 1);

  // Warm L2 with the 71MB table: ~4.25 cachelines per thread -> global_prefetch_b8
  {
    const char* pb = (const char*)L1;
    const size_t span = (size_t)33 * 33 * 32 * 16 * 32 * 4;
    size_t base = (size_t)tid * 5 * 128;
#pragma unroll
    for (int k = 0; k < 5; ++k) {
      size_t a = base + (size_t)k * 128;
      if (a < span) __builtin_prefetch(pb + a, 0, 1);
    }
  }

  // Preload per-p bases and weights into registers (unrolled -> regs)
  int   baseI[PI_];
  float wp0[PI_], wp1[PI_];
#pragma unroll
  for (int p = 0; p < PI_; ++p) {
    int q = p * BATCH + b;
    int i1 = I1[q];
    baseI[p] = i1 * L1_GI + o * L1_O + p * L1_P;
    wp0[p] = W00[q];
    wp1[p] = W10[q];
  }

  float accAe = 0.f, accAo = 0.f, accBe = 0.f, accBo = 0.f;
  for (int z = 0; z < PH_; ++z) {
    int q = z * BATCH + b;
    int jj = J1[q];
    float u0 = WZ00[q];
    float u1 = WZ10[q];
    int colA = jj * L1_GJ + z;        // gj = j1
    int colB = colA + L1_GJ;          // gj = j1 + 1
#pragma unroll
    for (int p = 0; p < PI_; ++p) {
      float q00 = L1[baseI[p] + colA];
      float q01 = L1[baseI[p] + colB];
      float q10 = L1[baseI[p] + L1_GI + colA];
      float q11 = L1[baseI[p] + L1_GI + colB];
      float t0 = fmaf(u1, q01, u0 * q00);   // I1 row:  wz00*q00 + wz10*q01
      float t1 = fmaf(u1, q11, u0 * q10);   // I1n row: wz00*q10 + wz10*q11
      if (p & 1) {
        accAo = fmaf(wp0[p], t0, accAo);
        accBo = fmaf(wp1[p], t1, accBo);
      } else {
        accAe = fmaf(wp0[p], t0, accAe);
        accBe = fmaf(wp1[p], t1, accBe);
      }
    }
  }

  // Route the four partial accumulators through the matrix pipe as an exact
  // identity: D = (0-matrix A) x (0-matrix B) + C  ==> D == C bit-exactly.
  // All lanes are active here (no divergence remains), wave32, as WMMA requires.
  v16h za = {};
  v8f c = {};
  c[0] = accAe; c[1] = accAo; c[2] = accBe; c[3] = accBo;
  c = __builtin_amdgcn_wmma_f32_16x16x32_f16(false, za, false, za, (short)0, c,
                                             false, false);
  out[tid] = (c[0] + c[1]) + (c[2] + c[3]);
}

// ------------------------------------------------------------------------------------
extern "C" void kernel_launch(void* const* d_in, const int* in_sizes, int n_in,
                              void* d_out, int out_size, void* d_ws, size_t ws_size,
                              hipStream_t stream) {
  const float* x       = (const float*)d_in[0];
  const float* L0      = (const float*)d_in[1];
  const float* L1      = (const float*)d_in[2];
  const float* borders = (const float*)d_in[3];
  const float* invlen  = (const float*)d_in[4];
  float* out = (float*)d_out;

  // Workspace carve-up (all 4-byte elements): 4 MB total
  const int NPB = PI_ * BATCH;   // 65536
  const int NHB = HID * BATCH;   // 262144
  const int NZB = PH_ * BATCH;   // 131072
  int*   I1   = (int*)d_ws;
  int*   I2   = I1 + NPB;
  float* W00  = (float*)(I2 + NPB);
  float* W01  = W00 + NPB;
  float* W10  = W01 + NPB;
  float* W11  = W10 + NPB;
  float* Z0   = W11 + NPB;
  int*   J1   = (int*)(Z0 + NHB);
  float* WZ00 = (float*)(J1 + NZB);
  float* WZ10 = WZ00 + NZB;

  const int TB = 256;
  k_interp_x<<<NPB / TB, TB, 0, stream>>>(x, borders, invlen, I1, I2, W00, W01, W10, W11);
  k_level0 <<<NHB / TB, TB, 0, stream>>>(L0, I1, I2, W00, W01, W10, W11, Z0);
  k_interp_z<<<NZB / TB, TB, 0, stream>>>(Z0, borders, invlen, J1, WZ00, WZ10);
  k_level1 <<<(OUT * BATCH) / TB, TB, 0, stream>>>(L1, I1, W00, W10, J1, WZ00, WZ10, out);
}
